// SVM_51462298140953
// MI455X (gfx1250) — matrix operand order repaired and checked by reading, not compiled, not run on hardware
//
#include <hip/hip_runtime.h>
#include <hip/hip_bf16.h>
#include <math.h>

typedef __attribute__((ext_vector_type(16))) __bf16 v16bf;
typedef __attribute__((ext_vector_type(8)))  float  v8f;

// Problem constants (from reference)
constexpr int Bsz = 2048;
constexpr int Nsz = 50000;
constexpr int Dsz = 512;

// Tiling
constexpr int BM      = 64;                    // x rows per block
constexpr int BN      = 32;                    // data rows per tile
constexpr int NCHUNKS = 64;                    // N-chunks (grid.y)
constexpr int NTILES  = (Nsz + BN - 1) / BN;   // 1563 (last tile ragged: coeff=0 pad)
constexpr int XPITCH  = Dsz + 8;               // halfs, padded vs 64-bank LDS
constexpr int DPITCH  = Dsz + 8;
constexpr int THREADS = 256;                   // 8 wave32
constexpr int BUFSTRIDE = BN * DPITCH;         // halfs per data buffer

__device__ __forceinline__ unsigned short f2bf(float f) {
  union { float f; unsigned u; } v; v.f = f;
  unsigned u = v.u;
  unsigned r = u + 0x7FFFu + ((u >> 16) & 1u);   // round-to-nearest-even
  return (unsigned short)(r >> 16);
}
__device__ __forceinline__ uint2 pack4(float4 v) {
  uint2 r;
  r.x = (unsigned)f2bf(v.x) | ((unsigned)f2bf(v.y) << 16);
  r.y = (unsigned)f2bf(v.z) | ((unsigned)f2bf(v.w) << 16);
  return r;
}

// ---------------------------------------------------------------------------
// Kernel 0: row norms + coeff = label * lambda.  One wave per row.
// ---------------------------------------------------------------------------
__global__ void __launch_bounds__(256)
svm_precompute(const float* __restrict__ x, const float* __restrict__ dset,
               const float* __restrict__ labels, const float* __restrict__ lam,
               float* __restrict__ x2, float* __restrict__ n2,
               float* __restrict__ coeff) {
  const int wid  = threadIdx.x >> 5;
  const int lane = threadIdx.x & 31;
  const int row  = blockIdx.x * 8 + wid;
  const float* src;
  if (row < Bsz)            src = x    + (size_t)row * Dsz;
  else if (row < Bsz + Nsz) src = dset + (size_t)(row - Bsz) * Dsz;
  else return;
  const float4* s4 = (const float4*)src;
  float s = 0.f;
#pragma unroll
  for (int i = 0; i < (Dsz / 4) / 32; ++i) {     // 4 float4 per lane
    float4 v = s4[lane + i * 32];
    s += v.x * v.x + v.y * v.y + v.z * v.z + v.w * v.w;
  }
  s += __shfl_xor(s, 16); s += __shfl_xor(s, 8); s += __shfl_xor(s, 4);
  s += __shfl_xor(s, 2);  s += __shfl_xor(s, 1);
  if (lane == 0) {
    if (row < Bsz) x2[row] = s;
    else { int n = row - Bsz; n2[n] = s; coeff[n] = labels[n] * lam[n]; }
  }
}

// ---------------------------------------------------------------------------
// Kernel 1: tiled bf16-WMMA GEMM fused with RBF kernel + N-partial reduction.
// Grid: (Bsz/BM) x NCHUNKS. Block: 256 thr = 8 waves in a 4(m) x 2(n) layout.
// ---------------------------------------------------------------------------
__global__ void __launch_bounds__(THREADS)
svm_rbf_main(const float* __restrict__ x, const float* __restrict__ dset,
             const float* __restrict__ x2, const float* __restrict__ n2,
             const float* __restrict__ coeff, float* __restrict__ partial) {
  extern __shared__ __align__(16) unsigned short smem[];
  unsigned short* xs    = smem;                  // BM x XPITCH bf16
  unsigned short* dbuf0 = xs + BM * XPITCH;      // 2 x (BN x DPITCH) bf16

  const int tid    = threadIdx.x;
  const int lane   = tid & 31;
  const int wid    = tid >> 5;
  const int m0     = (wid & 3) << 4;             // wave row offset   (0..48)
  const int wave_n = wid >> 2;                   // 0..1
  const int n0     = wave_n << 4;                // wave col offset
  const int hi     = lane >> 4;                  // half-wave group
  const int l15    = lane & 15;
  const int b_base = blockIdx.x * BM;
  const int chunk  = blockIdx.y;

  // ---- stage x tile: fp32 -> bf16 into LDS (once per block) ----
  {
    const float4* xg = (const float4*)(x + (size_t)b_base * Dsz);
#pragma unroll
    for (int i = 0; i < (BM * Dsz / 4) / THREADS; ++i) {   // 32 float4 / thread
      int j   = tid + i * THREADS;
      int row = j >> 7;                                    // 128 float4 per row
      int c4  = j & 127;
      float4 v = xg[(size_t)row * 128 + c4];
      *(uint2*)(xs + row * XPITCH + c4 * 4) = pack4(v);
    }
  }

  // per-lane ||x||^2 for the 8 C rows this lane owns
  float x2r[8];
#pragma unroll
  for (int r = 0; r < 8; ++r) x2r[r] = x2[b_base + m0 + hi * 8 + r];

  float sums[8];
#pragma unroll
  for (int r = 0; r < 8; ++r) sums[r] = 0.f;

  // ---- stage first data tile into buffer 0 ----
  int t = chunk;
  {
    int nb = t * BN;
#pragma unroll
    for (int i = 0; i < (BN * Dsz / 4) / THREADS; ++i) {   // 16 float4 / thread
      int j = tid + i * THREADS;
      int row = j >> 7, c4 = j & 127;
      int gr = nb + row; if (gr >= Nsz) gr = Nsz - 1;
      float4 v = ((const float4*)dset)[(size_t)gr * 128 + c4];
      *(uint2*)(dbuf0 + row * DPITCH + c4 * 4) = pack4(v);
    }
  }
  __syncthreads();

  // Double buffer toggled by an integer offset on the LDS pointer so the
  // compiler keeps addrspace(3) => ds_load_b128 (NOT flat) for B fragments.
  int bufOff = 0;
  for (; t < NTILES; t += NCHUNKS) {
    const int tn = t + NCHUNKS;
    const bool hasNext = tn < NTILES;

    // prefetch next tile into registers (overlaps with WMMA loop below;
    // these are LOADcnt-only and are not drained by the ds waits in the loop)
    uint2 pf[16];
    if (hasNext) {
      int nb = tn * BN;
#pragma unroll
      for (int i = 0; i < 16; ++i) {
        int j = tid + i * THREADS;
        int row = j >> 7, c4 = j & 127;
        int gr = nb + row; if (gr >= Nsz) gr = Nsz - 1;
        float4 v = ((const float4*)dset)[(size_t)gr * 128 + c4];
        pf[i] = pack4(v);
      }
    }

    // ---- WMMA K-loop: cross[16x16] += A(16x32 bf16) x B(32x16 bf16) ----
    const unsigned short* pa0 = xs + (size_t)(m0 + l15) * XPITCH + hi * 8;
    const unsigned short* pb0 =
        dbuf0 + bufOff + (size_t)(n0 + l15) * DPITCH + (hi << 3);
    v8f acc = {};
#pragma unroll 4
    for (int ks = 0; ks < Dsz; ks += 32) {
      union { v16bf v; uint4 q[2]; } A, Bm;
      // A layout: lane<16 holds K {ks..ks+7, ks+16..ks+23}; lane>=16 shifted by 8
      A.q[0]  = *(const uint4*)(pa0 + ks);
      A.q[1]  = *(const uint4*)(pa0 + ks + 16);
      // B layout: lane<16 holds col n, K {ks..ks+7, ks+16..ks+23}; lane>=16 shifted by 8
      Bm.q[0] = *(const uint4*)(pb0 + ks);
      Bm.q[1] = *(const uint4*)(pb0 + ks + 16);
      acc = __builtin_amdgcn_wmma_f32_16x16x32_bf16(
          false, A.v, false, Bm.v, (short)0, acc, false, false);
    }

    // ---- fused RBF: coeff * exp(-(x2 + n2 - 2*cross)/2), pad cols -> coeff=0
    const int nn = t * BN + n0 + l15;
    float cv = 0.f, nv = 0.f;
    if (nn < Nsz) { cv = coeff[nn]; nv = n2[nn]; }
#pragma unroll
    for (int r = 0; r < 8; ++r) {
      float d2 = x2r[r] + nv - 2.0f * acc[r];
      sums[r] += cv * __expf(-0.5f * d2);
    }

    // drain prefetch into the other buffer
    if (hasNext) {
      unsigned short* dn = dbuf0 + (bufOff ^ BUFSTRIDE);
#pragma unroll
      for (int i = 0; i < 16; ++i) {
        int j = tid + i * THREADS;
        int row = j >> 7, c4 = j & 127;
        *(uint2*)(dn + row * DPITCH + c4 * 4) = pf[i];
      }
    }
    __syncthreads();
    bufOff ^= BUFSTRIDE;
  }

  // ---- deterministic reduction over the 16 columns each half-wave holds ----
#pragma unroll
  for (int r = 0; r < 8; ++r) {
    float s = sums[r];
    s += __shfl_xor(s, 1);
    s += __shfl_xor(s, 2);
    s += __shfl_xor(s, 4);
    s += __shfl_xor(s, 8);
    sums[r] = s;   // all 16 lanes of the group now hold the row sum
  }
  float* red = (float*)smem;                     // reuse LDS: [2][BM] floats
  if (l15 == 0) {
#pragma unroll
    for (int r = 0; r < 8; ++r)
      red[wave_n * BM + m0 + hi * 8 + r] = sums[r];
  }
  __syncthreads();
  if (tid < BM) {
    partial[(size_t)chunk * Bsz + b_base + tid] = red[tid] + red[BM + tid];
  }
}

// ---------------------------------------------------------------------------
// Kernel 2: reduce chunk partials, add bias, tanh.
// ---------------------------------------------------------------------------
__global__ void __launch_bounds__(256)
svm_finalize(const float* __restrict__ partial, const float* __restrict__ bias,
             float* __restrict__ out) {
  int b = blockIdx.x * 256 + threadIdx.x;
  if (b >= Bsz) return;
  float s = 0.f;
#pragma unroll 8
  for (int c = 0; c < NCHUNKS; ++c) s += partial[(size_t)c * Bsz + b];
  out[b] = tanhf(s + bias[0]);
}

// ---------------------------------------------------------------------------
extern "C" void kernel_launch(void* const* d_in, const int* in_sizes, int n_in,
                              void* d_out, int out_size, void* d_ws, size_t ws_size,
                              hipStream_t stream) {
  const float* x      = (const float*)d_in[0];
  const float* dset   = (const float*)d_in[1];
  const float* labels = (const float*)d_in[2];
  const float* lam    = (const float*)d_in[3];
  const float* bias   = (const float*)d_in[4];
  float* out = (float*)d_out;

  // workspace carve-out (~0.93 MB)
  float* x2      = (float*)d_ws;            // [Bsz]
  float* n2      = x2 + Bsz;                // [Nsz]
  float* coeff   = n2 + Nsz;                // [Nsz]
  float* partial = coeff + Nsz;             // [NCHUNKS * Bsz]

  const int rows = Bsz + Nsz;
  svm_precompute<<<(rows + 7) / 8, 256, 0, stream>>>(x, dset, labels, lam,
                                                     x2, n2, coeff);

  dim3 grid(Bsz / BM, NCHUNKS);
  size_t smem = (size_t)(BM * XPITCH + 2 * BN * DPITCH) * sizeof(unsigned short); // 130 KB
  svm_rbf_main<<<grid, THREADS, smem, stream>>>(x, dset, x2, n2, coeff, partial);

  svm_finalize<<<(Bsz + 255) / 256, 256, 0, stream>>>(partial, bias, out);
}
